// TemporalGraphNetwork_34033320854155
// MI455X (gfx1250) — compile-verified, matrix operand order checked by reading, not run on hardware
//
#include <hip/hip_runtime.h>

// ---------------------------------------------------------------------------
// TemporalGraphNetwork forward for MI455X (gfx1250, wave32, WMMA)
// ---------------------------------------------------------------------------

#define NN    50000      // nodes
#define EE    200000     // edges
#define TOTE  250000     // edges + self loops
#define SRT   262144     // sort size (pow2 >= EE)
#define MPAD  50048      // nodes padded to 128-row macro tiles (391*128)
#define BATCH 200
#define NSTEP 1000       // EE / BATCH

typedef __attribute__((ext_vector_type(16))) _Float16 v16h;
typedef __attribute__((ext_vector_type(8)))  _Float16 v8h;
typedef __attribute__((ext_vector_type(8)))  float    v8f;

__device__ inline v8f wmma16(v16h a, v16h b, v8f c) {
  // D = A(16x32 f16) * B(32x16 f16) + C(16x16 f32)
  return __builtin_amdgcn_wmma_f32_16x16x32_f16(
      /*neg_a=*/false, a, /*neg_b=*/false, b,
      /*c_mod=*/(short)0, c, /*reuse_a=*/false, /*reuse_b=*/false);
}

// A fragment: 16x32 (MxK) f16, row-major, leading dim ld (halves).
// ISA layout: lane m=l&15, kb=8*(l>>4); elems 0..7 -> K=kb..kb+7 (contiguous),
// elems 8..15 -> K=kb+16..kb+23 (contiguous) => two aligned 16B vector loads.
__device__ inline v16h frag_a(const _Float16* base, int ld) {
  int lane = threadIdx.x & 31;
  int m = lane & 15, kb = (lane >> 4) << 3;
  const _Float16* p = base + (size_t)m * ld + kb;
  v8h lo = *(const v8h*)(p);
  v8h hi = *(const v8h*)(p + 16);
  return __builtin_shufflevector(lo, hi, 0, 1, 2, 3, 4, 5, 6, 7, 8, 9, 10, 11,
                                 12, 13, 14, 15);
}

// B fragment from pre-packed weights: layout [chunk][ntile][lane][16 halves],
// one aligned 32B load per lane.
__device__ inline v16h frag_bp(const _Float16* Bp, int c, int nt, int ni) {
  return *(const v16h*)(Bp +
                        (((size_t)c * nt + ni) * 32 + (threadIdx.x & 31)) * 16);
}

// C/D layout: VGPR r, lane l -> row = r + 8*(l>>4), col = l&15
__device__ inline void tile_store_f16(_Float16* C, int ldc, int mi, int ni,
                                      v8f acc, const float* bias, int relu) {
  int lane = threadIdx.x & 31;
  int n = lane & 15, mh = (lane >> 4) << 3;
  float b = bias ? bias[ni * 16 + n] : 0.f;
#pragma unroll
  for (int r = 0; r < 8; ++r) {
    float v = acc[r] + b;
    if (relu) v = fmaxf(v, 0.f);
    C[(size_t)(mi * 16 + mh + r) * ldc + ni * 16 + n] = (_Float16)v;
  }
}

__device__ inline void tile_store_f32(float* C, int ldc, int mi, int ni,
                                      v8f acc, const float* bias, int relu) {
  int lane = threadIdx.x & 31;
  int n = lane & 15, mh = (lane >> 4) << 3;
  float b = bias ? bias[ni * 16 + n] : 0.f;
#pragma unroll
  for (int r = 0; r < 8; ++r) {
    float v = acc[r] + b;
    if (relu) v = fmaxf(v, 0.f);
    C[(size_t)(mi * 16 + mh + r) * ldc + ni * 16 + n] = v;
  }
}

__device__ inline unsigned f2ord(float f) {
  unsigned u = __float_as_uint(f);
  return (u & 0x80000000u) ? ~u : (u | 0x80000000u);
}
__device__ inline float ord2f(unsigned u) {
  u = (u & 0x80000000u) ? (u & 0x7FFFFFFFu) : ~u;
  return __uint_as_float(u);
}
__device__ inline float sigm(float x) { return 1.f / (1.f + __expf(-x)); }

// ---------------------------------------------------------------------------
// Sort: float time keys -> ordered uint, padded bitonic sort (ascending)
// ---------------------------------------------------------------------------
__global__ void prep_sort(const float* times, unsigned* keys, unsigned* vals) {
  int i = blockIdx.x * blockDim.x + threadIdx.x;
  if (i >= SRT) return;
  if (i < EE) { keys[i] = f2ord(times[i]); vals[i] = (unsigned)i; }
  else        { keys[i] = 0xFFFFFFFFu;     vals[i] = 0u; }
}

__global__ void bitonic_step(unsigned* keys, unsigned* vals, int j, int k) {
  int i = blockIdx.x * blockDim.x + threadIdx.x;
  int p = i ^ j;
  if (p <= i || p >= SRT) return;
  unsigned a = keys[i], b = keys[p];
  bool asc = ((i & k) == 0);
  if (asc ? (a > b) : (a < b)) {
    keys[i] = b; keys[p] = a;
    unsigned t = vals[i]; vals[i] = vals[p]; vals[p] = t;
  }
}

// ---------------------------------------------------------------------------
// Weight packing: f32 -> f16 in WMMA B-fragment layout
//   dst[((c*nt + ni)*32 + lane)*16 + i]  holds  B[k][n] with
//   k = c*32 + 8*(lane>>4) + (i<8 ? i : i+8),  n = ni*16 + (lane&15)
// ---------------------------------------------------------------------------
__device__ inline void unpack_idx(int idx, int nt, int& k, int& n) {
  int i = idx & 15;
  int lane = (idx >> 4) & 31;
  int rest = idx >> 9;
  int ni = rest % nt, c = rest / nt;
  k = c * 32 + ((lane >> 4) << 3) + (i < 8 ? i : i + 8);
  n = ni * 16 + (lane & 15);
}
// B[k][n] = src[k*Nt + n]
__global__ void pack_b_direct(_Float16* dst, const float* src, int Kt, int Nt) {
  int idx = blockIdx.x * blockDim.x + threadIdx.x;
  if (idx >= Kt * Nt) return;
  int k, n;
  unpack_idx(idx, Nt >> 4, k, n);
  dst[idx] = (_Float16)src[(size_t)k * Nt + n];
}
// B[k][n] = src[n*Kt + k]   (gru W.T)
__global__ void pack_b_trans(_Float16* dst, const float* src, int Kt, int Nt) {
  int idx = blockIdx.x * blockDim.x + threadIdx.x;
  if (idx >= Kt * Nt) return;
  int k, n;
  unpack_idx(idx, Nt >> 4, k, n);
  dst[idx] = (_Float16)src[(size_t)n * Kt + k];
}
// B[k][h*32+j] = src[h][k][j], src [4][IN][32], Nt=128
__global__ void pack_b_gat(_Float16* dst, const float* src, int IN) {
  int idx = blockIdx.x * blockDim.x + threadIdx.x;
  if (idx >= IN * 128) return;
  int k, n;
  unpack_idx(idx, 8, k, n);
  dst[idx] = (_Float16)src[((size_t)(n >> 5) * IN + k) * 32 + (n & 31)];
}

// ---------------------------------------------------------------------------
// Temporal memory scan: single persistent workgroup (8 wave32)
// ---------------------------------------------------------------------------
__global__ __launch_bounds__(256) void scan_kernel(
    const float* nodef, const float* edge_attr, const float* etimes,
    const unsigned* svals, const int* eidx0,
    const float* time_w, const float* time_b,
    const _Float16* W1p, const float* b1,
    const _Float16* W2p, const float* b2,
    const _Float16* WihTp, const float* bih,
    const _Float16* WhhTp, const float* bhh,
    float* mem,
    _Float16* m_in,   // [208][320]
    _Float16* msgA,   // [208][192]
    _Float16* msgB,   // [208][192]
    _Float16* h16,    // [208][128]
    float* h32,       // [208][128]
    float* gi,        // [208][384]
    float* gh)        // [208][384]
{
  __shared__ int s_src[208];
  __shared__ unsigned char s_last[208];
  const int tid = threadIdx.x;
  const int wave = tid >> 5;

  for (int step = 0; step < NSTEP; ++step) {
    // ---- phase 1a: source ids ----
    if (tid < 208) {
      int s = -1;
      if (tid < 200) s = eidx0[svals[step * BATCH + tid]];
      s_src[tid] = s;
    }
    __syncthreads();
    // ---- phase 1b: gather m_in = [x_src | h | edge_attr | sin(w t + b)] ----
    if (tid < 208) {
      int row = tid, s = s_src[row];
      unsigned char last = 0;
      if (row < 200) {
        last = 1;
        for (int j = row + 1; j < 200; ++j)
          if (s_src[j] == s) { last = 0; break; }
      }
      s_last[row] = last;
      if (row < 200) {
        unsigned e = svals[step * BATCH + row];
        float t = etimes[e];
        for (int c = 0; c < 128; ++c) {
          float nf = nodef[(size_t)s * 128 + c];
          float hm = mem[(size_t)s * 128 + c];
          m_in[(size_t)row * 320 + c]       = (_Float16)nf;
          m_in[(size_t)row * 320 + 128 + c] = (_Float16)hm;
          h16[(size_t)row * 128 + c] = (_Float16)hm;
          h32[(size_t)row * 128 + c] = hm;
        }
        for (int c = 0; c < 32; ++c) {
          m_in[(size_t)row * 320 + 256 + c] = (_Float16)edge_attr[(size_t)e * 32 + c];
          m_in[(size_t)row * 320 + 288 + c] = (_Float16)__sinf(time_w[c] * t + time_b[c]);
        }
      } else {
        for (int c = 0; c < 320; ++c) m_in[(size_t)row * 320 + c] = (_Float16)0.f;
        for (int c = 0; c < 128; ++c) {
          h16[(size_t)row * 128 + c] = (_Float16)0.f;
          h32[(size_t)row * 128 + c] = 0.f;
        }
      }
    }
    __syncthreads();

    // ---- GEMM1: relu(m_in[208x320] @ W1[320x192] + b1) -> msgA (f16) ----
    for (int tile = wave; tile < 156; tile += 8) {   // 13 x 12 tiles
      int mi = tile / 12, ni = tile % 12;
      v8f acc = {};
#pragma unroll
      for (int kk = 0; kk < 10; ++kk)
        acc = wmma16(frag_a(m_in + (size_t)mi * 16 * 320 + kk * 32, 320),
                     frag_bp(W1p, kk, 12, ni), acc);
      tile_store_f16(msgA, 192, mi, ni, acc, b1, 1);
    }
    __syncthreads();

    // ---- GEMM2: msgA[208x192] @ W2[192x192] + b2 -> msgB (f16) ----
    for (int tile = wave; tile < 156; tile += 8) {
      int mi = tile / 12, ni = tile % 12;
      v8f acc = {};
#pragma unroll
      for (int kk = 0; kk < 6; ++kk)
        acc = wmma16(frag_a(msgA + (size_t)mi * 16 * 192 + kk * 32, 192),
                     frag_bp(W2p, kk, 12, ni), acc);
      tile_store_f16(msgB, 192, mi, ni, acc, b2, 0);
    }
    __syncthreads();

    // ---- GEMM3/4: gh = h @ Whh.T + bhh ; gi = msgB @ Wih.T + bih ----
    for (int job = wave; job < 624; job += 8) {      // 2 * (13 x 24) tiles
      if (job < 312) {
        int mi = job / 24, ni = job % 24;
        v8f acc = {};
#pragma unroll
        for (int kk = 0; kk < 4; ++kk)
          acc = wmma16(frag_a(h16 + (size_t)mi * 16 * 128 + kk * 32, 128),
                       frag_bp(WhhTp, kk, 24, ni), acc);
        tile_store_f32(gh, 384, mi, ni, acc, bhh, 0);
      } else {
        int t = job - 312;
        int mi = t / 24, ni = t % 24;
        v8f acc = {};
#pragma unroll
        for (int kk = 0; kk < 6; ++kk)
          acc = wmma16(frag_a(msgB + (size_t)mi * 16 * 192 + kk * 32, 192),
                       frag_bp(WihTp, kk, 24, ni), acc);
        tile_store_f32(gi, 384, mi, ni, acc, bih, 0);
      }
    }
    __syncthreads();

    // ---- GRU gates + scatter (last occurrence wins) ----
    for (int i = tid; i < 200 * 128; i += 256) {
      int row = i >> 7, c = i & 127;
      float ir = gi[(size_t)row * 384 + c];
      float iz = gi[(size_t)row * 384 + 128 + c];
      float in_ = gi[(size_t)row * 384 + 256 + c];
      float hr = gh[(size_t)row * 384 + c];
      float hz = gh[(size_t)row * 384 + 128 + c];
      float hn_ = gh[(size_t)row * 384 + 256 + c];
      float r = sigm(ir + hr);
      float z = sigm(iz + hz);
      float cg = tanhf(in_ + r * hn_);
      float h = h32[(size_t)row * 128 + c];
      float out = (1.f - z) * cg + z * h;
      if (s_last[row]) mem[(size_t)s_src[row] * 128 + c] = out;
    }
    __syncthreads();
  }
}

// ---------------------------------------------------------------------------
// x = [node_features | memory | sin(w*t_max + b)]  as f16, padded rows = 0
// ---------------------------------------------------------------------------
__global__ void build_x(const float* nodef, const float* mem,
                        const float* tw, const float* tb,
                        const unsigned* keys, _Float16* x16) {
  long long idx = (long long)blockIdx.x * blockDim.x + threadIdx.x;
  if (idx >= (long long)MPAD * 288) return;
  int row = (int)(idx / 288), col = (int)(idx % 288);
  float v = 0.f;
  if (row < NN) {
    if (col < 128)      v = nodef[(size_t)row * 128 + col];
    else if (col < 256) v = mem[(size_t)row * 128 + (col - 128)];
    else {
      float tmax = ord2f(keys[EE - 1]);
      v = __sinf(tw[col - 256] * tmax + tb[col - 256]);
    }
  }
  x16[idx] = (_Float16)v;
}

// ---------------------------------------------------------------------------
// Tiled WMMA GEMM: C[MPAD x Nn] = A[MPAD x Ka](f16) * Bp(packed f16) (+bias,relu)
// block = 256 threads computes a 128 x Nn macro tile; A panels staged in LDS,
// B fragments read directly from pre-packed weights (hot in WGP$/L2).
// ---------------------------------------------------------------------------
__global__ __launch_bounds__(256) void gemm_f16(const _Float16* A,
                                                const _Float16* Bp, float* C,
                                                const float* bias, int Ka,
                                                int Nn, int relu) {
  __shared__ __align__(16) _Float16 sA[128 * 32];
  const int wave = threadIdx.x >> 5;
  const int lane = threadIdx.x & 31;
  const int nt = Nn >> 4;
  v8f acc[8] = {};
  const _Float16* ga = A + (size_t)blockIdx.x * 128 * Ka;

  for (int kk = 0; kk < Ka; kk += 32) {
    // stage A panel 128x32 (uint4 = 8 halves per chunk)
    for (int c = threadIdx.x; c < 512; c += 256) {
      int row = c >> 2, seg = c & 3;
      *(uint4*)(sA + row * 32 + seg * 8) =
          *(const uint4*)(ga + (size_t)row * Ka + kk + seg * 8);
    }
    // prefetch next A panel into caches (global_prefetch_b8)
    if (kk + 32 < Ka && threadIdx.x < 128)
      __builtin_prefetch(ga + (size_t)threadIdx.x * Ka + kk + 32, 0, 1);
    __syncthreads();

    v16h af = frag_a(sA + wave * 16 * 32, 32);
    int c32 = kk >> 5;
#pragma unroll
    for (int ni = 0; ni < 8; ++ni)
      if (ni < nt) acc[ni] = wmma16(af, frag_bp(Bp, c32, nt, ni), acc[ni]);
    __syncthreads();
  }

  int n = lane & 15, mh = (lane >> 4) << 3;
  for (int ni = 0; ni < nt; ++ni) {
    float b = bias ? bias[ni * 16 + n] : 0.f;
#pragma unroll
    for (int r = 0; r < 8; ++r) {
      size_t row = (size_t)blockIdx.x * 128 + wave * 16 + mh + r;
      float v = acc[ni][r] + b;
      if (relu) v = fmaxf(v, 0.f);
      C[row * Nn + ni * 16 + n] = v;
    }
  }
}

// ---------------------------------------------------------------------------
// GAT attention pipeline
// ---------------------------------------------------------------------------
__global__ void esed_kernel(const float* xp, const float* as, const float* ad,
                            float* es, float* ed) {
  int nidx = blockIdx.x * blockDim.x + threadIdx.x;
  if (nidx >= NN) return;
  float s0 = 0, s1 = 0, s2 = 0, s3 = 0, d0 = 0, d1 = 0, d2 = 0, d3 = 0;
  const float* xr = xp + (size_t)nidx * 128;
#pragma unroll 4
  for (int j = 0; j < 32; ++j) {
    s0 += xr[j] * as[j];            d0 += xr[j] * ad[j];
    s1 += xr[32 + j] * as[32 + j];  d1 += xr[32 + j] * ad[32 + j];
    s2 += xr[64 + j] * as[64 + j];  d2 += xr[64 + j] * ad[64 + j];
    s3 += xr[96 + j] * as[96 + j];  d3 += xr[96 + j] * ad[96 + j];
  }
  es[nidx] = s0; es[NN + nidx] = s1; es[2 * NN + nidx] = s2; es[3 * NN + nidx] = s3;
  ed[nidx] = d0; ed[NN + nidx] = d1; ed[2 * NN + nidx] = d2; ed[3 * NN + nidx] = d3;
}

__global__ void att_max(const float* es, const float* ed, const int* e0,
                        const int* e1, float* elog, unsigned* segmax) {
  int i = blockIdx.x * blockDim.x + threadIdx.x;
  if (i >= TOTE) return;
  int s = (i < EE) ? e0[i] : (i - EE);
  int d = (i < EE) ? e1[i] : (i - EE);
#pragma unroll
  for (int h = 0; h < 4; ++h) {
    float v = es[h * NN + s] + ed[h * NN + d];
    v = (v > 0.f) ? v : 0.2f * v;                 // leaky_relu(0.2)
    elog[(size_t)h * TOTE + i] = v;
    atomicMax(segmax + h * NN + d, f2ord(v));
  }
}

__global__ void att_expsum(const int* e1, float* elog, const unsigned* segmax,
                           float* den) {
  int i = blockIdx.x * blockDim.x + threadIdx.x;
  if (i >= TOTE) return;
  int d = (i < EE) ? e1[i] : (i - EE);
#pragma unroll
  for (int h = 0; h < 4; ++h) {
    float m = ord2f(segmax[h * NN + d]);
    float ex = __expf(elog[(size_t)h * TOTE + i] - m);
    elog[(size_t)h * TOTE + i] = ex;
    atomicAdd(den + h * NN + d, ex);
  }
}

__global__ void att_scatter(const int* e0, const int* e1, const float* elog,
                            const float* den, const float* xp, float* agg) {
  int t = blockIdx.x * blockDim.x + threadIdx.x;
  if (t >= 4 * TOTE) return;
  int h = t / TOTE, i = t - h * TOTE;
  int s = (i < EE) ? e0[i] : (i - EE);
  int d = (i < EE) ? e1[i] : (i - EE);
  float alpha = elog[t] / den[h * NN + d];
  const float* xr = xp + (size_t)s * 128 + h * 32;
  float* orow = agg + (size_t)d * 128 + h * 32;
#pragma unroll
  for (int c = 0; c < 32; ++c) atomicAdd(orow + c, alpha * xr[c]);
}

__global__ void gat_finalize(const float* agg, const float* bias,
                             _Float16* out16, int relu) {
  long long idx = (long long)blockIdx.x * blockDim.x + threadIdx.x;
  if (idx >= (long long)MPAD * 128) return;
  int row = (int)(idx >> 7), col = (int)(idx & 127);
  float v = 0.f;
  if (row < NN) {
    v = agg[idx] + bias[col];
    if (relu) v = fmaxf(v, 0.f);
  }
  out16[idx] = (_Float16)v;
}

// ---------------------------------------------------------------------------
// Classifier tail: logits[n] = h[n,:64] . W2 + b2
// ---------------------------------------------------------------------------
__global__ void cls_out(const float* hbuf, const float* w2, const float* b2,
                        float* out) {
  int n = blockIdx.x * blockDim.x + threadIdx.x;
  if (n >= NN) return;
  const float* hr = hbuf + (size_t)n * 64;
  float s = 0.f;
#pragma unroll 8
  for (int c = 0; c < 64; ++c) s += hr[c] * w2[c];
  out[n] = s + b2[0];
}

// ---------------------------------------------------------------------------
// Host launch
// ---------------------------------------------------------------------------
extern "C" void kernel_launch(void* const* d_in, const int* in_sizes, int n_in,
                              void* d_out, int out_size, void* d_ws,
                              size_t ws_size, hipStream_t stream) {
  (void)in_sizes; (void)n_in; (void)out_size; (void)ws_size;

  const float* nodef   = (const float*)d_in[0];
  const float* eattr   = (const float*)d_in[1];
  const float* etimes  = (const float*)d_in[2];
  const float* time_w  = (const float*)d_in[3];
  const float* time_b  = (const float*)d_in[4];
  const float* msg_W1  = (const float*)d_in[5];
  const float* msg_b1  = (const float*)d_in[6];
  const float* msg_W2  = (const float*)d_in[7];
  const float* msg_b2  = (const float*)d_in[8];
  const float* gru_Wih = (const float*)d_in[9];
  const float* gru_Whh = (const float*)d_in[10];
  const float* gru_bih = (const float*)d_in[11];
  const float* gru_bhh = (const float*)d_in[12];
  const float* gat1_W  = (const float*)d_in[13];
  const float* gat1_as = (const float*)d_in[14];
  const float* gat1_ad = (const float*)d_in[15];
  const float* gat1_b  = (const float*)d_in[16];
  const float* gat2_W  = (const float*)d_in[17];
  const float* gat2_as = (const float*)d_in[18];
  const float* gat2_ad = (const float*)d_in[19];
  const float* gat2_b  = (const float*)d_in[20];
  const float* cls_W1  = (const float*)d_in[21];
  const float* cls_b1  = (const float*)d_in[22];
  const float* cls_W2  = (const float*)d_in[23];
  const float* cls_b2  = (const float*)d_in[24];
  const int*   eidx0   = (const int*)d_in[25];
  const int*   eidx1   = eidx0 + EE;
  float* out = (float*)d_out;

  // ---- workspace bump allocator ----
  char* wp = (char*)d_ws;
  auto alloc = [&](size_t bytes) -> void* {
    void* r = (void*)wp;
    wp += (bytes + 255) & ~(size_t)255;
    return r;
  };
  unsigned* keys  = (unsigned*)alloc(SRT * 4);
  unsigned* vals  = (unsigned*)alloc(SRT * 4);
  float* mem      = (float*)alloc((size_t)NN * 128 * 4);
  _Float16* x16   = (_Float16*)alloc((size_t)MPAD * 288 * 2);
  float* xp       = (float*)alloc((size_t)MPAD * 128 * 4);
  float* agg      = (float*)alloc((size_t)MPAD * 128 * 4);
  _Float16* x2_16 = (_Float16*)alloc((size_t)MPAD * 128 * 2);
  _Float16* xf16  = (_Float16*)alloc((size_t)MPAD * 128 * 2);
  float* hbuf     = (float*)alloc((size_t)MPAD * 64 * 4);
  float* es       = (float*)alloc((size_t)4 * NN * 4);
  float* ed       = (float*)alloc((size_t)4 * NN * 4);
  float* elog     = (float*)alloc((size_t)4 * TOTE * 4);
  unsigned* segmx = (unsigned*)alloc((size_t)4 * NN * 4);
  float* den      = (float*)alloc((size_t)4 * NN * 4);
  _Float16* W1p   = (_Float16*)alloc(320 * 192 * 2);
  _Float16* W2p   = (_Float16*)alloc(192 * 192 * 2);
  _Float16* WihTp = (_Float16*)alloc(192 * 384 * 2);
  _Float16* WhhTp = (_Float16*)alloc(128 * 384 * 2);
  _Float16* Wg1p  = (_Float16*)alloc(288 * 128 * 2);
  _Float16* Wg2p  = (_Float16*)alloc(128 * 128 * 2);
  _Float16* Wc1p  = (_Float16*)alloc(128 * 64 * 2);
  _Float16* m_in  = (_Float16*)alloc(208 * 320 * 2);
  _Float16* msgA  = (_Float16*)alloc(208 * 192 * 2);
  _Float16* msgB  = (_Float16*)alloc(208 * 192 * 2);
  _Float16* h16   = (_Float16*)alloc(208 * 128 * 2);
  float* h32      = (float*)alloc(208 * 128 * 4);
  float* gi       = (float*)alloc(208 * 384 * 4);
  float* gh       = (float*)alloc(208 * 384 * 4);

  const int T = 256;
  auto G = [](long long n) { return (unsigned)((n + 255) / 256); };

  // memory starts at zero each call
  hipMemsetAsync(mem, 0, (size_t)NN * 128 * 4, stream);

  // 1) sort edges by time (bitonic over ordered-uint keys)
  prep_sort<<<G(SRT), T, 0, stream>>>(etimes, keys, vals);
  for (int k = 2; k <= SRT; k <<= 1)
    for (int j = k >> 1; j > 0; j >>= 1)
      bitonic_step<<<G(SRT), T, 0, stream>>>(keys, vals, j, k);

  // 2) weights -> f16 WMMA fragment layouts
  pack_b_direct<<<G(320 * 192), T, 0, stream>>>(W1p, msg_W1, 320, 192);
  pack_b_direct<<<G(192 * 192), T, 0, stream>>>(W2p, msg_W2, 192, 192);
  pack_b_direct<<<G(128 * 64), T, 0, stream>>>(Wc1p, cls_W1, 128, 64);
  pack_b_trans<<<G(192 * 384), T, 0, stream>>>(WihTp, gru_Wih, 192, 384);
  pack_b_trans<<<G(128 * 384), T, 0, stream>>>(WhhTp, gru_Whh, 128, 384);
  pack_b_gat<<<G(288 * 128), T, 0, stream>>>(Wg1p, gat1_W, 288);
  pack_b_gat<<<G(128 * 128), T, 0, stream>>>(Wg2p, gat2_W, 128);

  // 3) sequential temporal-memory scan (persistent single workgroup)
  scan_kernel<<<1, 256, 0, stream>>>(
      nodef, eattr, etimes, vals, eidx0, time_w, time_b,
      W1p, msg_b1, W2p, msg_b2, WihTp, gru_bih, WhhTp, gru_bhh,
      mem, m_in, msgA, msgB, h16, h32, gi, gh);

  // 4) x = [nf | mem | gte]
  build_x<<<G((long long)MPAD * 288), T, 0, stream>>>(nodef, mem, time_w,
                                                      time_b, keys, x16);

  // ---- GAT layer 1 ----
  gemm_f16<<<MPAD / 128, 256, 0, stream>>>(x16, Wg1p, xp, nullptr, 288, 128, 0);
  esed_kernel<<<G(NN), T, 0, stream>>>(xp, gat1_as, gat1_ad, es, ed);
  hipMemsetAsync(segmx, 0, (size_t)4 * NN * 4, stream);
  hipMemsetAsync(den, 0, (size_t)4 * NN * 4, stream);
  hipMemsetAsync(agg, 0, (size_t)MPAD * 128 * 4, stream);
  att_max<<<G(TOTE), T, 0, stream>>>(es, ed, eidx0, eidx1, elog, segmx);
  att_expsum<<<G(TOTE), T, 0, stream>>>(eidx1, elog, segmx, den);
  att_scatter<<<G((long long)4 * TOTE), T, 0, stream>>>(eidx0, eidx1, elog, den,
                                                        xp, agg);
  gat_finalize<<<G((long long)MPAD * 128), T, 0, stream>>>(agg, gat1_b, x2_16, 1);

  // ---- GAT layer 2 ----
  gemm_f16<<<MPAD / 128, 256, 0, stream>>>(x2_16, Wg2p, xp, nullptr, 128, 128, 0);
  esed_kernel<<<G(NN), T, 0, stream>>>(xp, gat2_as, gat2_ad, es, ed);
  hipMemsetAsync(segmx, 0, (size_t)4 * NN * 4, stream);
  hipMemsetAsync(den, 0, (size_t)4 * NN * 4, stream);
  hipMemsetAsync(agg, 0, (size_t)MPAD * 128 * 4, stream);
  att_max<<<G(TOTE), T, 0, stream>>>(es, ed, eidx0, eidx1, elog, segmx);
  att_expsum<<<G(TOTE), T, 0, stream>>>(eidx1, elog, segmx, den);
  att_scatter<<<G((long long)4 * TOTE), T, 0, stream>>>(eidx0, eidx1, elog, den,
                                                        xp, agg);
  gat_finalize<<<G((long long)MPAD * 128), T, 0, stream>>>(agg, gat2_b, xf16, 0);

  // ---- classifier ----
  gemm_f16<<<MPAD / 128, 256, 0, stream>>>(xf16, Wc1p, hbuf, cls_b1, 128, 64, 1);
  cls_out<<<G(NN), T, 0, stream>>>(hbuf, cls_W2, cls_b2, out);
}
